// MultiGranularityGlobalPooling_43499428774418
// MI455X (gfx1250) — compile-verified, matrix operand order checked by reading, not run on hardware
//
#include <hip/hip_runtime.h>

// ---------------- problem constants ----------------
constexpr int E  = 32768;
constexpr int H  = 256;
constexpr int S  = 32;
constexpr int NH = 8;
constexpr int B  = 64;
constexpr int HD = 32;          // H / NH
constexpr int SH = S * H;       // 8192

typedef __attribute__((ext_vector_type(16))) __bf16 v16bf;
typedef __attribute__((ext_vector_type(8)))  float  v8f;

// ---------------- workspace layout (bytes) ----------------
constexpr size_t OFF_EDGEBF = 0;                                   // [E][H] bf16 row-major
constexpr size_t OFF_CSW    = OFF_EDGEBF + (size_t)E * H * 2;      // Cmat swizzled  [8][16][512]
constexpr size_t OFF_WVSW   = OFF_CSW    + (size_t)H * H * 2;      // WvT swizzled   [8][16][512]
constexpr size_t OFF_WOSW   = OFF_WVSW   + (size_t)H * H * 2;      // WoT swizzled   [8][16][512]
constexpr size_t OFF_W1SW   = OFF_WOSW   + (size_t)H * H * 2;      // W1T swizzled   [256][16][512]
constexpr size_t OFF_Q      = OFF_W1SW   + (size_t)SH * H * 2;     // [S][H]  f32
constexpr size_t OFF_SCORES = OFF_Q      + (size_t)S * H * 4;      // [E][H]  f32 row-major
constexpr size_t OFF_VSW    = OFF_SCORES + (size_t)E * H * 4;      // V swizzled [E/32][16][512] bf16
constexpr size_t OFF_ATTN   = OFF_VSW    + (size_t)E * H * 2;      // [B][S][H] bf16 row-major
constexpr size_t OFF_GR     = OFF_ATTN   + (size_t)B * S * H * 2;  // [B][S][H] bf16 (= flat [B][SH])
constexpr size_t OFF_H1     = OFF_GR     + (size_t)B * S * H * 2;  // [B][H]  bf16
constexpr size_t OFF_WT     = OFF_H1     + (size_t)B * H * 2;      // wT [H][E] bf16 (weights, transposed)
constexpr size_t OFF_MB     = OFF_WT     + (size_t)H * E * 2;      // [B][H] f32 segment max
constexpr size_t OFF_RD     = OFF_MB     + (size_t)B * H * 4;      // [B][H] f32 1/denom
constexpr size_t OFF_SEG    = OFF_RD     + (size_t)B * H * 4;      // [B][2] int

// ---------------- async-to-LDS (guarded; falls back to plain copies) ----------
#if __has_builtin(__builtin_amdgcn_global_load_async_to_lds_b128)
#define HAVE_ASYNC_LDS 1
#endif

typedef int ai4 __attribute__((vector_size(16)));
typedef __attribute__((address_space(1))) ai4* g_ai4_ptr;
typedef __attribute__((address_space(3))) ai4* l_ai4_ptr;

__device__ __forceinline__ void cp16(__bf16* l, const __bf16* g) {
#if defined(HAVE_ASYNC_LDS)
  __builtin_amdgcn_global_load_async_to_lds_b128((g_ai4_ptr)g, (l_ai4_ptr)l, 0, 0);
#else
  *(uint4*)l = *(const uint4*)g;
#endif
}
__device__ __forceinline__ void async_wait() {
#if defined(HAVE_ASYNC_LDS)
#if __has_builtin(__builtin_amdgcn_s_wait_asynccnt)
  __builtin_amdgcn_s_wait_asynccnt(0);
#else
  asm volatile("s_wait_asynccnt 0x0" ::: "memory");
#endif
#endif
}

// ---------------- small helpers ----------------
__device__ __forceinline__ __bf16 f2bf(float x) { return (__bf16)x; }

union FragU { v16bf v; uint4 q[2]; __bf16 h[16]; };
union Pk8   { __bf16 h[8]; uint4 q; };

__device__ __forceinline__ v8f zero8() {
  v8f z;
#pragma unroll
  for (int i = 0; i < 8; ++i) z[i] = 0.0f;
  return z;
}

__device__ __forceinline__ v8f wmma_bf16(v16bf a, v16bf b, v8f c) {
  return __builtin_amdgcn_wmma_f32_16x16x32_bf16(
      false, a, false, b, (short)0, c, false, false);
}

// A fragment 16x32 (MxK) bf16 from row-major source: two contiguous b128 loads.
__device__ __forceinline__ v16bf ldfragA(const __bf16* p, int ld, int m0, int k0, int lane) {
  const int m = m0 + (lane & 15);
  const int g = lane >> 4;
  const __bf16* r = p + (size_t)m * ld + k0 + (g << 3);
  FragU u;
  u.q[0] = *(const uint4*)r;
  u.q[1] = *(const uint4*)(r + 16);
  return u.v;
}

// B fragment 32x16 (KxN) bf16 from swizzled tile (512 bf16, fragment-major).
__device__ __forceinline__ v16bf ldfragBsw(const __bf16* tile, int lane) {
  const __bf16* r = tile + lane * 16;
  FragU u;
  u.q[0] = *(const uint4*)r;
  u.q[1] = *(const uint4*)(r + 8);
  return u.v;
}

// ================= prep kernels =================

__global__ void k_prep_q(const float* __restrict__ seed, const float* __restrict__ Wq,
                         float* __restrict__ q) {
  const int s = blockIdx.x, j = threadIdx.x;
  float acc = 0.f;
  for (int h = 0; h < H; ++h) acc += seed[s * H + h] * Wq[j * H + h];
  q[s * H + j] = acc;
}

__global__ void k_prep_cmat_sw(const float* __restrict__ Wk, const float* __restrict__ q,
                               __bf16* __restrict__ Csw) {
  const size_t u = (size_t)blockIdx.x * blockDim.x + threadIdx.x;
  const int i    = (int)(u & 15);
  const int lane = (int)((u >> 4) & 31);
  const int nt   = (int)((u >> 9) & 15);
  const int kc   = (int)(u >> 13);
  const int g = lane >> 4;
  const int K = kc * 32 + (i & 7) + ((i >> 3) << 4) + (g << 3);   // = h
  const int N = nt * 16 + (lane & 15);                            // = nh*32 + s
  const int nh = N >> 5, s = N & 31;
  float acc = 0.f;
  for (int d = 0; d < HD; ++d)
    acc += Wk[(nh * HD + d) * H + K] * q[s * H + nh * HD + d];
  Csw[u] = f2bf(acc * 0.17677669529663687f); // 1/sqrt(32)
}

__global__ void k_swz(const float* __restrict__ W, __bf16* __restrict__ out, int ldW) {
  const size_t u = (size_t)blockIdx.x * blockDim.x + threadIdx.x;
  const int i    = (int)(u & 15);
  const int lane = (int)((u >> 4) & 31);
  const int nt   = (int)((u >> 9) & 15);
  const size_t kc = u >> 13;
  const int g = lane >> 4;
  const int K = (int)kc * 32 + (i & 7) + ((i >> 3) << 4) + (g << 3);
  const int N = nt * 16 + (lane & 15);
  out[u] = f2bf(W[(size_t)N * ldW + K]);
}

__global__ void k_cvt8(const float* __restrict__ src, __bf16* __restrict__ dst, int n8) {
  const int idx = blockIdx.x * blockDim.x + threadIdx.x;
  if (idx >= n8) return;
  const float4* s = (const float4*)src;
  const float4 a = s[idx * 2], b = s[idx * 2 + 1];
  Pk8 p;
  p.h[0] = f2bf(a.x); p.h[1] = f2bf(a.y); p.h[2] = f2bf(a.z); p.h[3] = f2bf(a.w);
  p.h[4] = f2bf(b.x); p.h[5] = f2bf(b.y); p.h[6] = f2bf(b.z); p.h[7] = f2bf(b.w);
  ((uint4*)dst)[idx] = p.q;
}

// ================= fused scores + V projection =================
__global__ __launch_bounds__(256) void k_scores_v(
    const __bf16* __restrict__ edge_bf, const __bf16* __restrict__ Csw,
    const __bf16* __restrict__ Wvsw, float* __restrict__ scores,
    __bf16* __restrict__ Vsw) {
  __shared__ __attribute__((aligned(16))) __bf16 Asm[64 * 256];   // 32 KB
  __shared__ __attribute__((aligned(16))) __bf16 Bss[16 * 512];   // 16 KB
  __shared__ __attribute__((aligned(16))) __bf16 Bsv[16 * 512];   // 16 KB

  const int tid  = threadIdx.x;
  const int lane = tid & 31;
  const int wave = tid >> 5;
  const int row0 = blockIdx.x * 64;

  for (int i = tid; i < 64 * 256 / 8; i += 256)
    cp16(Asm + i * 8, edge_bf + (size_t)row0 * H + i * 8);

  const int m0  = (wave & 3) * 16;
  const int nt0 = (wave >> 2) * 8;

  v8f acc_s[8], acc_v[8];
#pragma unroll
  for (int t = 0; t < 8; ++t) { acc_s[t] = zero8(); acc_v[t] = zero8(); }

  for (int kc = 0; kc < 8; ++kc) {
    __syncthreads();
    for (int i = tid; i < 16 * 512 / 8; i += 256) {
      cp16(Bss + i * 8, Csw  + (size_t)kc * 16 * 512 + i * 8);
      cp16(Bsv + i * 8, Wvsw + (size_t)kc * 16 * 512 + i * 8);
    }
    if (kc < 7) {
      __builtin_prefetch(Csw  + (size_t)(kc + 1) * 16 * 512 + tid * 32, 0, 1);
      __builtin_prefetch(Wvsw + (size_t)(kc + 1) * 16 * 512 + tid * 32, 0, 1);
    }
    async_wait();
    __syncthreads();

    const v16bf a = ldfragA(Asm, H, m0, kc * 32, lane);
    // software-pipelined B fragment loads
    v16bf bs = ldfragBsw(Bss + nt0 * 512, lane);
    v16bf bv = ldfragBsw(Bsv + nt0 * 512, lane);
#pragma unroll
    for (int t = 0; t < 8; ++t) {
      const int tn = (t + 1) & 7;
      const v16bf bs_n = ldfragBsw(Bss + (nt0 + tn) * 512, lane);
      const v16bf bv_n = ldfragBsw(Bsv + (nt0 + tn) * 512, lane);
      acc_s[t] = wmma_bf16(a, bs, acc_s[t]);
      acc_v[t] = wmma_bf16(a, bv, acc_v[t]);
      bs = bs_n; bv = bv_n;
    }
  }

  const int g = lane >> 4, nlo = lane & 15;
#pragma unroll
  for (int t = 0; t < 8; ++t)
#pragma unroll
    for (int r = 0; r < 8; ++r) {
      const int e = row0 + m0 + r + g * 8;
      const int j = (nt0 + t) * 16 + nlo;
      scores[(size_t)e * H + j] = acc_s[t][r];
      const int c   = e >> 5, kin = e & 31;
      const int ii  = (kin & 7) | (((kin >> 4) & 1) << 3);
      const int gg  = (kin >> 3) & 1;
      const size_t idx = ((((size_t)c * 16 + (j >> 4)) * 32) + gg * 16 + (j & 15)) * 16 + ii;
      Vsw[idx] = f2bf(acc_v[t][r]);
    }
}

// ================= segment reduce: per-(graph, channel) max & 1/denom ========
__global__ __launch_bounds__(256) void k_seg_reduce(
    const float* __restrict__ scores, const int* __restrict__ batch,
    float* __restrict__ mbuf, float* __restrict__ rdbuf, int* __restrict__ segbuf) {
  __shared__ int seg[2];
  const int b = blockIdx.x, tid = threadIdx.x;

  if (tid == 0) {
    int lo = 0, hi = E;
    while (lo < hi) { int mid = (lo + hi) >> 1; if (batch[mid] < b) lo = mid + 1; else hi = mid; }
    seg[0] = lo;
    int lo2 = lo, hi2 = E;
    while (lo2 < hi2) { int mid = (lo2 + hi2) >> 1; if (batch[mid] < b + 1) lo2 = mid + 1; else hi2 = mid; }
    seg[1] = lo2;
    segbuf[b * 2]     = seg[0];
    segbuf[b * 2 + 1] = seg[1];
  }
  __syncthreads();
  const int start = seg[0], end = seg[1];

  const int col = tid;
  float vmax = -3.4e38f;
  for (int e = start; e < end; ++e) vmax = fmaxf(vmax, scores[(size_t)e * H + col]);
  float den = 0.f;
  for (int e = start; e < end; ++e) den += __expf(scores[(size_t)e * H + col] - vmax);
  mbuf[b * H + col]  = vmax;
  rdbuf[b * H + col] = (den > 0.f) ? (1.f / den) : 0.f;
}

// ================= elementwise softmax weights, stored transposed ============
__global__ __launch_bounds__(256) void k_wexp(
    const float* __restrict__ scores, const int* __restrict__ batch,
    const float* __restrict__ mbuf, const float* __restrict__ rdbuf,
    __bf16* __restrict__ wT) {
  const int e = blockIdx.x, col = threadIdx.x;
  const int b = batch[e];
  const float w = __expf(scores[(size_t)e * H + col] - mbuf[b * H + col]) * rdbuf[b * H + col];
  wT[(size_t)col * E + e] = f2bf(w);
}

// ================= attend: attended = wT(32xEg per head) x V(Egx32) ==========
__global__ __launch_bounds__(256) void k_attend(
    const __bf16* __restrict__ wT, const __bf16* __restrict__ Vsw,
    const int* __restrict__ segbuf, __bf16* __restrict__ attn) {
  const int b = blockIdx.x;
  const int tid = threadIdx.x, lane = tid & 31, nh = tid >> 5;
  const int start = segbuf[b * 2], end = segbuf[b * 2 + 1];

  v8f acc[2][2];
#pragma unroll
  for (int mt = 0; mt < 2; ++mt)
#pragma unroll
    for (int nt = 0; nt < 2; ++nt) acc[mt][nt] = zero8();

  const int g = lane >> 4;
  const __bf16* wTh = wT + (size_t)(nh * 32) * E;

  for (int e0 = start & ~31; e0 < end; e0 += 32) {
    const int c = e0 >> 5;
    const bool bdry = (e0 < start) || (e0 + 32 > end);
    v16bf afr[2], bfr[2];
#pragma unroll
    for (int mt = 0; mt < 2; ++mt) {
      v16bf a = ldfragA(wTh, E, mt * 16, e0, lane);
      if (bdry) {
#pragma unroll
        for (int i = 0; i < 16; ++i) {
          const int e = e0 + (i & 7) + ((i >> 3) << 4) + (g << 3);
          if (e < start || e >= end) a[i] = f2bf(0.f);
        }
      }
      afr[mt] = a;
    }
#pragma unroll
    for (int nt = 0; nt < 2; ++nt)
      bfr[nt] = ldfragBsw(Vsw + ((size_t)c * 16 + (nh * 2 + nt)) * 512, lane);
#pragma unroll
    for (int mt = 0; mt < 2; ++mt)
#pragma unroll
      for (int nt = 0; nt < 2; ++nt)
        acc[mt][nt] = wmma_bf16(afr[mt], bfr[nt], acc[mt][nt]);
  }

  const int nlo = lane & 15;
#pragma unroll
  for (int mt = 0; mt < 2; ++mt)
#pragma unroll
    for (int nt = 0; nt < 2; ++nt)
#pragma unroll
      for (int r = 0; r < 8; ++r) {
        const int s  = mt * 16 + r + g * 8;
        const int hd = nt * 16 + nlo;
        attn[((size_t)b * S + s) * H + nh * 32 + hd] = f2bf(acc[mt][nt][r]);
      }
}

// ================= out projection: graph_repr = attn @ WoT + bo ==============
__global__ __launch_bounds__(256) void k_outproj(
    const __bf16* __restrict__ Abf, const __bf16* __restrict__ Bsw,
    const float* __restrict__ bias, __bf16* __restrict__ Obf) {
  __shared__ __attribute__((aligned(16))) __bf16 Asm[64 * 256];
  __shared__ __attribute__((aligned(16))) __bf16 Bsm[16 * 512];

  const int tid = threadIdx.x, lane = tid & 31, wave = tid >> 5;
  const int row0 = blockIdx.x * 64;

  for (int i = tid; i < 64 * 256 / 8; i += 256)
    cp16(Asm + i * 8, Abf + (size_t)row0 * H + i * 8);

  const int m0  = (wave & 3) * 16;
  const int nt0 = (wave >> 2) * 8;

  v8f acc[8];
#pragma unroll
  for (int t = 0; t < 8; ++t) acc[t] = zero8();

  for (int kc = 0; kc < 8; ++kc) {
    __syncthreads();
    for (int i = tid; i < 16 * 512 / 8; i += 256)
      cp16(Bsm + i * 8, Bsw + (size_t)kc * 16 * 512 + i * 8);
    async_wait();
    __syncthreads();

    const v16bf a = ldfragA(Asm, H, m0, kc * 32, lane);
    v16bf bf = ldfragBsw(Bsm + nt0 * 512, lane);
#pragma unroll
    for (int t = 0; t < 8; ++t) {
      const int tn = (t + 1) & 7;
      const v16bf bn = ldfragBsw(Bsm + (nt0 + tn) * 512, lane);
      acc[t] = wmma_bf16(a, bf, acc[t]);
      bf = bn;
    }
  }

  const int g = lane >> 4, nlo = lane & 15;
#pragma unroll
  for (int t = 0; t < 8; ++t)
#pragma unroll
    for (int r = 0; r < 8; ++r) {
      const int m = row0 + m0 + r + g * 8;
      const int n = (nt0 + t) * 16 + nlo;
      Obf[(size_t)m * H + n] = f2bf(acc[t][r] + bias[n]);
    }
}

// ================= MLP1: h1 = silu(flat @ W1T + b1) =================
__global__ __launch_bounds__(256) void k_mlp1(
    const __bf16* __restrict__ flat, const __bf16* __restrict__ W1sw,
    const float* __restrict__ b1, __bf16* __restrict__ h1) {
  __shared__ float red[64 * 16];

  const int tid = threadIdx.x, lane = tid & 31, wave = tid >> 5;
  const int ntb = blockIdx.x;

  for (int i = tid; i < 64 * 16; i += 256) red[i] = 0.f;
  __syncthreads();

  v8f acc[4];
#pragma unroll
  for (int mt = 0; mt < 4; ++mt) acc[mt] = zero8();

  const int kc0 = wave * 32;
  v16bf bfrag = ldfragBsw(W1sw + ((size_t)kc0 * 16 + ntb) * 512, lane);
  for (int kc = kc0; kc < kc0 + 32; ++kc) {
    const int k0 = kc * 32;
    const int kn = (kc + 1 < kc0 + 32) ? kc + 1 : kc0;
    const v16bf bnext = ldfragBsw(W1sw + ((size_t)kn * 16 + ntb) * 512, lane);
    v16bf afr[4];
#pragma unroll
    for (int mt = 0; mt < 4; ++mt) afr[mt] = ldfragA(flat, SH, mt * 16, k0, lane);
#pragma unroll
    for (int mt = 0; mt < 4; ++mt) acc[mt] = wmma_bf16(afr[mt], bfrag, acc[mt]);
    bfrag = bnext;
  }

  const int g = lane >> 4, nlo = lane & 15;
#pragma unroll
  for (int mt = 0; mt < 4; ++mt)
#pragma unroll
    for (int r = 0; r < 8; ++r) {
      const int m = mt * 16 + r + g * 8;
      atomicAdd(&red[m * 16 + nlo], acc[mt][r]);
    }
  __syncthreads();

  for (int i = tid; i < 64 * 16; i += 256) {
    const int m = i >> 4, n = i & 15;
    const float x = red[i] + b1[ntb * 16 + n];
    const float s = x / (1.f + __expf(-x));
    h1[m * H + ntb * 16 + n] = f2bf(s);
  }
}

// ================= MLP2: out = h1 @ W2T + b2 (tiny, scalar) =================
__global__ void k_mlp2(const __bf16* __restrict__ h1, const float* __restrict__ W2,
                       const float* __restrict__ b2, float* __restrict__ out) {
  const int b = blockIdx.x, j = threadIdx.x;
  float acc = 0.f;
  for (int h = 0; h < H; ++h) acc += (float)h1[b * H + h] * W2[j * H + h];
  out[b * H + j] = acc + b2[j];
}

// ================= launcher =================
extern "C" void kernel_launch(void* const* d_in, const int* in_sizes, int n_in,
                              void* d_out, int out_size, void* d_ws, size_t ws_size,
                              hipStream_t stream) {
  const float* edge  = (const float*)d_in[0];
  const int*   batch = (const int*)d_in[2];
  const float* seed  = (const float*)d_in[3];
  const float* Wq = (const float*)d_in[4];
  const float* Wk = (const float*)d_in[5];
  const float* Wv = (const float*)d_in[6];
  const float* Wo = (const float*)d_in[7];
  const float* bo = (const float*)d_in[8];
  const float* W1 = (const float*)d_in[9];
  const float* b1 = (const float*)d_in[10];
  const float* W2 = (const float*)d_in[11];
  const float* b2 = (const float*)d_in[12];
  float* out = (float*)d_out;

  char* ws = (char*)d_ws;
  __bf16* edge_bf = (__bf16*)(ws + OFF_EDGEBF);
  __bf16* Csw     = (__bf16*)(ws + OFF_CSW);
  __bf16* Wvsw    = (__bf16*)(ws + OFF_WVSW);
  __bf16* Wosw    = (__bf16*)(ws + OFF_WOSW);
  __bf16* W1sw    = (__bf16*)(ws + OFF_W1SW);
  float*  qbuf    = (float*)(ws + OFF_Q);
  float*  scores  = (float*)(ws + OFF_SCORES);
  __bf16* Vsw     = (__bf16*)(ws + OFF_VSW);
  __bf16* attn    = (__bf16*)(ws + OFF_ATTN);
  __bf16* grbf    = (__bf16*)(ws + OFF_GR);
  __bf16* h1bf    = (__bf16*)(ws + OFF_H1);
  __bf16* wT      = (__bf16*)(ws + OFF_WT);
  float*  mbuf    = (float*)(ws + OFF_MB);
  float*  rdbuf   = (float*)(ws + OFF_RD);
  int*    segbuf  = (int*)(ws + OFF_SEG);

  // prep
  k_prep_q<<<S, H, 0, stream>>>(seed, Wq, qbuf);
  k_prep_cmat_sw<<<H, H, 0, stream>>>(Wk, qbuf, Csw);
  k_cvt8<<<(E * H / 8 + 255) / 256, 256, 0, stream>>>(edge, edge_bf, E * H / 8);
  k_swz<<<H, H, 0, stream>>>(Wv, Wvsw, H);
  k_swz<<<H, H, 0, stream>>>(Wo, Wosw, H);
  k_swz<<<SH, H, 0, stream>>>(W1, W1sw, SH);

  // fused scores + V projection
  k_scores_v<<<E / 64, 256, 0, stream>>>(edge_bf, Csw, Wvsw, scores, Vsw);

  // segment softmax (split for full-chip parallelism)
  k_seg_reduce<<<B, 256, 0, stream>>>(scores, batch, mbuf, rdbuf, segbuf);
  k_wexp<<<E, 256, 0, stream>>>(scores, batch, mbuf, rdbuf, wT);
  k_attend<<<B, 256, 0, stream>>>(wT, Vsw, segbuf, attn);

  // out projection
  k_outproj<<<(B * S) / 64, 256, 0, stream>>>(attn, Wosw, bo, grbf);

  // MLP
  k_mlp1<<<H / 16, 256, 0, stream>>>(grbf, W1sw, b1, h1bf);
  k_mlp2<<<B, H, 0, stream>>>(h1bf, W2, b2, out);
}
